// cross_set_score_515396076024
// MI455X (gfx1250) — compile-verified, matrix operand order checked by reading, not run on hardware
//
#include <hip/hip_runtime.h>

// ---------------------------------------------------------------------------
// cross_set_score for MI455X (gfx1250, wave32, WMMA f16 16x16x32)
//
// Pipeline:
//   k0: W1 (f32 256x256) -> f16, pre-swizzled into WMMA B-fragment layout
//   k1: per (i,j): h = x[i,j] @ W1 via v_wmma_f32_16x16x32_f16, stored f16
//       in WMMA fragment layout (serves as both A and B operand in stage 2)
//   k2: per (j,i): sum(relu(h[j,i,h] @ h[i,j,h]^T / 8)) per head, then
//       normalize by nItem and mix heads with W2.
// ---------------------------------------------------------------------------

typedef __attribute__((ext_vector_type(16))) _Float16 v16h;
typedef __attribute__((ext_vector_type(8)))  _Float16 v8h;
typedef __attribute__((ext_vector_type(4)))  _Float16 v4h;
typedef __attribute__((ext_vector_type(8)))  float    v8f;

static constexpr int N_  = 64;    // set grid dims
static constexpr int M_  = 64;    // items per set
static constexpr int F_  = 256;   // input features
static constexpr int H_  = 4;     // heads
static constexpr int D_  = 64;    // head size
static constexpr int HD_ = H_ * D_;            // 256
static constexpr int FRAG_HALFS = 32 * 16;     // 512 halfs per 16x32 fragment
static constexpr int W1F_HALFS  = F_ * HD_;    // 65536
static constexpr int HEAD_HALFS = 4 * 2 * FRAG_HALFS; // [Mt=4][Kt=2] frags = 4096
static constexpr int PAIR_HALFS = H_ * HEAD_HALFS;    // 16384
static constexpr int LDS_STRIDE = 264;         // 256 halfs + 8 pad (kills bank conflicts)

// --------------------------------------------------------------------------
// Kernel 0: W1 f32 -> f16, swizzled to fragment layout [Nt(16)][Kt(8)][lane(32)][16]
// Per ISA 7.12.2 (16-bit operand, K=32): lanes 0-15 hold K {0..7, 16..23},
// lanes 16-31 hold K {8..15, 24..31}; lane&15 selects the N (or M) index.
// --------------------------------------------------------------------------
__global__ void swizzle_w1_kernel(const float* __restrict__ W1,
                                  _Float16* __restrict__ w1f) {
  int id   = blockIdx.x * blockDim.x + threadIdx.x;   // 0 .. 65535
  int jj   = id & 15;
  int lane = (id >> 4) & 31;
  int kt   = (id >> 9) & 7;
  int nt   = (id >> 12) & 15;
  int n    = nt * 16 + (lane & 15);
  int koff = (lane >> 4) * 8;
  int k    = kt * 32 + koff + ((jj < 8) ? jj : (8 + jj));  // jj>=8 -> +16+(jj-8)
  w1f[id] = (_Float16)W1[k * HD_ + n];
}

// --------------------------------------------------------------------------
// Kernel 1: projection. One block per (i,j); 8 waves compute C[64x256] =
// X[64x256] @ W1[256x256] with WMMA, then emit f16 fragments of h.
// Wave w covers Mtiles {2*(w>>2), 2*(w>>2)+1} x Ntiles {4*(w&3)..+3}.
// --------------------------------------------------------------------------
__global__ void __launch_bounds__(256)
proj_kernel(const float* __restrict__ x,
            const _Float16* __restrict__ w1f,
            _Float16* __restrict__ hbuf) {
  __shared__ __align__(16) _Float16 lds[64 * LDS_STRIDE];

  const int t    = threadIdx.x;
  const int lane = t & 31;
  const int wave = t >> 5;
  const int bi   = blockIdx.y;
  const int bj   = blockIdx.x;

  // ---- stage X[i,j] (64x256 f32) into LDS as f16, coalesced float4 loads ----
  {
    const float4* x4 = (const float4*)(x + (size_t)(bi * N_ + bj) * (M_ * F_));
#pragma unroll
    for (int k = 0; k < 16; ++k) {
      int f4  = k * 256 + t;          // 4096 float4 total
      float4 v = x4[f4];
      int row = f4 >> 6;              // /64 float4 per row
      int col = (f4 & 63) * 4;
      v4h hv = { (_Float16)v.x, (_Float16)v.y, (_Float16)v.z, (_Float16)v.w };
      *(v4h*)&lds[row * LDS_STRIDE + col] = hv;
    }
  }
  __syncthreads();

  const int mtBase = (wave >> 2) * 2;       // 0 or 2
  const int ntBase = (wave & 3) * 4;        // 0,4,8,12
  const int mrow0  = mtBase * 16 + (lane & 15);
  const int klane  = (lane >> 4) << 3;      // 0 or 8

  v8f acc[2][4] = {};

#pragma unroll
  for (int kt = 0; kt < 8; ++kt) {
    // A fragments from LDS (padded stride -> conflict-free b128 reads)
    v16h a[2];
#pragma unroll
    for (int mi = 0; mi < 2; ++mi) {
      const _Float16* src = &lds[(mrow0 + mi * 16) * LDS_STRIDE + kt * 32 + klane];
      v8h lo = *(const v8h*)src;
      v8h hi = *(const v8h*)(src + 16);
      v16h av;
#pragma unroll
      for (int u = 0; u < 8; ++u) { av[u] = lo[u]; av[8 + u] = hi[u]; }
      a[mi] = av;
    }
    // B fragments straight from pre-swizzled W1 (32B/lane, fully coalesced)
#pragma unroll
    for (int ni = 0; ni < 4; ++ni) {
      v16h b = *(const v16h*)(w1f + ((ntBase + ni) * 8 + kt) * FRAG_HALFS + lane * 16);
#pragma unroll
      for (int mi = 0; mi < 2; ++mi) {
        acc[mi][ni] = __builtin_amdgcn_wmma_f32_16x16x32_f16(
            false, a[mi], false, b, (short)0, acc[mi][ni], false, false);
      }
    }
  }
  __syncthreads();   // all waves done reading X; reuse LDS for C

  // ---- spill C tiles to LDS as f16 (C layout: vgpr r, lane l -> m=r+8*(l>>4)) ----
#pragma unroll
  for (int mi = 0; mi < 2; ++mi) {
#pragma unroll
    for (int ni = 0; ni < 4; ++ni) {
      int n = (ntBase + ni) * 16 + (lane & 15);
#pragma unroll
      for (int r = 0; r < 8; ++r) {
        int m = (mtBase + mi) * 16 + r + ((lane >> 4) << 3);
        lds[m * LDS_STRIDE + n] = (_Float16)acc[mi][ni][r];
      }
    }
  }
  __syncthreads();

  // ---- emit h fragments: [head][Mt][Kt][lane][16], coalesced 32B/lane ----
  _Float16* dst = hbuf + (size_t)(bi * N_ + bj) * PAIR_HALFS;
#pragma unroll
  for (int it = 0; it < 4; ++it) {
    int c     = it * 256 + t;       // 1024 chunks
    int lane2 = c & 31;
    int ktc   = (c >> 5) & 1;
    int mtc   = (c >> 6) & 3;
    int head  = (c >> 8) & 3;
    int m     = mtc * 16 + (lane2 & 15);
    int kb    = ktc * 32 + ((lane2 >> 4) << 3);
    const _Float16* src = &lds[m * LDS_STRIDE + head * 64 + kb];
    v8h lo = *(const v8h*)src;
    v8h hi = *(const v8h*)(src + 16);
    v8h* o = (v8h*)(dst + (((head * 4 + mtc) * 2 + ktc) * FRAG_HALFS) + lane2 * 16);
    o[0] = lo;
    o[1] = hi;
  }
}

// --------------------------------------------------------------------------
// Kernel 2: pairwise scores. One block per (p=j, q=i); 2 waves per head.
// S = relu((h[p,q,head] @ h[q,p,head]^T) / 8) summed over all 64x64 entries.
// Since both operands are row-sets of h and the op is A*B^T, the A and B
// fragment layouts coincide -> both read the same pre-swizzled buffer.
// --------------------------------------------------------------------------
__global__ void __launch_bounds__(256)
score_kernel(const _Float16* __restrict__ hbuf,
             const float* __restrict__ nItem,
             const float* __restrict__ W2,
             float* __restrict__ out) {
  __shared__ float ssum[H_];
  const int t    = threadIdx.x;
  const int lane = t & 31;
  const int wave = t >> 5;
  const int p    = blockIdx.y;   // j
  const int q    = blockIdx.x;   // i

  if (t < H_) ssum[t] = 0.0f;
  __syncthreads();

  const int head   = wave >> 1;
  const int mtBase = (wave & 1) * 2;
  const _Float16* Abase = hbuf + ((size_t)(p * N_ + q) * H_ + head) * HEAD_HALFS;
  const _Float16* Bbase = hbuf + ((size_t)(q * N_ + p) * H_ + head) * HEAD_HALFS;

  v8f acc[2][4] = {};
#pragma unroll
  for (int kt = 0; kt < 2; ++kt) {
    v16h a[2], b[4];
#pragma unroll
    for (int mi = 0; mi < 2; ++mi)
      a[mi] = *(const v16h*)(Abase + ((mtBase + mi) * 2 + kt) * FRAG_HALFS + lane * 16);
#pragma unroll
    for (int ni = 0; ni < 4; ++ni)
      b[ni] = *(const v16h*)(Bbase + (ni * 2 + kt) * FRAG_HALFS + lane * 16);
#pragma unroll
    for (int mi = 0; mi < 2; ++mi)
#pragma unroll
      for (int ni = 0; ni < 4; ++ni)
        acc[mi][ni] = __builtin_amdgcn_wmma_f32_16x16x32_f16(
            false, a[mi], false, b[ni], (short)0, acc[mi][ni], false, false);
  }

  // fused relu + sum (scale by 1/sqrt(D)=0.125 commutes past relu & sum)
  float s = 0.0f;
#pragma unroll
  for (int mi = 0; mi < 2; ++mi)
#pragma unroll
    for (int ni = 0; ni < 4; ++ni)
#pragma unroll
      for (int r = 0; r < 8; ++r)
        s += fmaxf(acc[mi][ni][r], 0.0f);

#pragma unroll
  for (int off = 16; off > 0; off >>= 1)
    s += __shfl_xor(s, off, 32);
  if (lane == 0) atomicAdd(&ssum[head], s);
  __syncthreads();

  if (t == 0) {
    float inv = 0.125f / (nItem[p] * nItem[q]);
    float r = 0.0f;
#pragma unroll
    for (int h = 0; h < H_; ++h) r += ssum[h] * W2[h];
    out[p * N_ + q] = r * inv;
  }
}

// --------------------------------------------------------------------------
extern "C" void kernel_launch(void* const* d_in, const int* in_sizes, int n_in,
                              void* d_out, int out_size, void* d_ws, size_t ws_size,
                              hipStream_t stream) {
  const float* x     = (const float*)d_in[0];
  const float* nItem = (const float*)d_in[1];
  const float* W1    = (const float*)d_in[2];
  const float* W2    = (const float*)d_in[3];
  float* out = (float*)d_out;

  _Float16* w1f  = (_Float16*)d_ws;            // 128 KB, L2-resident
  _Float16* hbuf = w1f + W1F_HALFS;            // 134 MB f16 fragments

  swizzle_w1_kernel<<<256, 256, 0, stream>>>(W1, w1f);
  proj_kernel<<<dim3(N_, N_), 256, 0, stream>>>(x, w1f, hbuf);
  score_kernel<<<dim3(N_, N_), 256, 0, stream>>>(hbuf, nItem, W2, out);
}
